// VectorQuantizer_51410758533496
// MI455X (gfx1250) — matrix operand order repaired and checked by reading, not compiled, not run on hardware
//
#include <hip/hip_runtime.h>
#include <hip/hip_bf16.h>
#include <math.h>

// ---------------------------------------------------------------------------
// VQ-VAE vector quantizer for MI455X (gfx1250, wave32, WMMA bf16)
//   inputs   [N=32768, D=768] fp32
//   codebook [K=4096,  D=768] fp32
// Heavy op: argmin_k ( ||e_k||^2 - 2 x.e_k )  == 32768x4096x768 GEMM + argmin
//
// v3: B tile staged with GLOBAL_LOAD_ASYNC_TO_LDS_B128 (ASYNCcnt), double
//     buffered in LDS so the global->LDS transfer of tile t+1 overlaps the
//     24 WMMAs of tile t.  A fragments stay register-resident.
// ---------------------------------------------------------------------------

#define D_DIM   768
#define D_PAD   776                 // padded LDS row stride (dword bank skew 4)
#define K_CB    4096
#define N_TOK   32768
#define CTILE   16                  // codebook rows per WMMA column tile
#define NTILES  (K_CB / CTILE)      // 256
#define KSTEPS  (D_DIM / 32)        // 24 wmma k-steps over D

typedef __bf16 v16bf __attribute__((ext_vector_type(16)));
typedef __bf16 v8bf  __attribute__((ext_vector_type(8)));
typedef float  v8f   __attribute__((ext_vector_type(8)));
typedef float  v4f   __attribute__((ext_vector_type(4)));

union V16U { v16bf v; v8bf h[2]; };

// ---------------------------------------------------------------------------
// K1: codebook fp32 -> bf16, and e2[k] = sum(e_k^2).  One block per code row.
// ---------------------------------------------------------------------------
__global__ void __launch_bounds__(128) vq_prep_codebook(
    const float* __restrict__ cb, __bf16* __restrict__ cb_bf,
    float* __restrict__ e2)
{
    const int k = blockIdx.x;
    const int tid = threadIdx.x;
    float acc = 0.f;
    for (int j = tid; j < D_DIM; j += 128) {
        float f = cb[(size_t)k * D_DIM + j];
        cb_bf[(size_t)k * D_DIM + j] = (__bf16)f;
        acc += f * f;
    }
    #pragma unroll
    for (int m = 16; m >= 1; m >>= 1) acc += __shfl_xor(acc, m, 32);
    __shared__ float ws[4];
    if ((tid & 31) == 0) ws[tid >> 5] = acc;
    __syncthreads();
    if (tid == 0) e2[k] = ws[0] + ws[1] + ws[2] + ws[3];
}

// ---------------------------------------------------------------------------
// K2: inputs fp32 -> bf16 (float4 -> 4x bf16, 64-bit stores)
// ---------------------------------------------------------------------------
__global__ void __launch_bounds__(256) vq_prep_inputs(
    const float* __restrict__ x, __bf16* __restrict__ x_bf, int n4)
{
    int i = blockIdx.x * 256 + threadIdx.x;
    if (i >= n4) return;
    v4f f = ((const v4f*)x)[i];
    __bf16 o[4] = { (__bf16)f.x, (__bf16)f.y, (__bf16)f.z, (__bf16)f.w };
    *(uint64_t*)(x_bf + (size_t)i * 4) = *(const uint64_t*)o;
}

// ---------------------------------------------------------------------------
// Async global -> LDS copy of one thread's 96B share of a B tile.
// Per ISA 08_async_tensor: inst offset is added to BOTH the LDS address
// (VDST vgpr) and the global address (SADDR + VADDR + offset).
// ---------------------------------------------------------------------------
__device__ __forceinline__ void async_copy96(unsigned lds_off, unsigned gvoff,
                                             const __bf16* sbase) {
    asm volatile(
        "global_load_async_to_lds_b128 %0, %1, %2 offset:0\n\t"
        "global_load_async_to_lds_b128 %0, %1, %2 offset:16\n\t"
        "global_load_async_to_lds_b128 %0, %1, %2 offset:32\n\t"
        "global_load_async_to_lds_b128 %0, %1, %2 offset:48\n\t"
        "global_load_async_to_lds_b128 %0, %1, %2 offset:64\n\t"
        "global_load_async_to_lds_b128 %0, %1, %2 offset:80"
        :: "v"(lds_off), "v"(gvoff), "s"(sbase) : "memory");
}

__device__ __forceinline__ void wait_async0() {
    asm volatile("s_wait_asynccnt 0" ::: "memory");
}

// ---------------------------------------------------------------------------
// K3: fused GEMM + argmin.  Block = 256 thr = 8 waves; wave owns 16 token rows.
// A fragments in VGPRs; B tile double-buffered in LDS via async loads.
// ---------------------------------------------------------------------------
#define LDS_B_ELEMS (CTILE * D_PAD)            // one buffer, elements
#define LDS_B_BYTES (LDS_B_ELEMS * 2)          // 24832 B
#define LDS_BYTES   (2 * LDS_B_BYTES + K_CB * 4)

__global__ void __launch_bounds__(256, 1) vq_argmin_kernel(
    const __bf16* __restrict__ Xbf, const __bf16* __restrict__ Ebf,
    const float* __restrict__ e2g, int* __restrict__ out_idx)
{
    extern __shared__ char smem[];                       // dynamic LDS @ off 0
    __bf16* Bs  = (__bf16*)smem;                         // 2 x (16 x D_PAD)
    float*  E2s = (float*)(smem + 2 * LDS_B_BYTES);      // K_CB floats

    const int tid     = threadIdx.x;
    const int wave    = tid >> 5;
    const int lane    = tid & 31;
    const int half    = lane >> 4;     // 0: lanes 0-15, 1: lanes 16-31
    const int l15     = lane & 15;
    const int rowbase = blockIdx.x * 128;

    // Stage e2 into LDS (16KB)
    for (int c = tid; c < K_CB; c += 256) E2s[c] = e2g[c];

    // Load all 24 A fragments for this lane's matrix row into registers.
    // A-matrix 16x32 bf16 layout: lane holds row l15; per k-step two 8-elem
    // runs at columns ks*32 + 8*half and ks*32 + 16 + 8*half.
    V16U afrag[KSTEPS];
    {
        const __bf16* arow = Xbf + (size_t)(rowbase + wave * 16 + l15) * D_DIM;
        #pragma unroll
        for (int ks = 0; ks < KSTEPS; ++ks) {
            afrag[ks].h[0] = *(const v8bf*)(arow + ks * 32 + 8 * half);
            afrag[ks].h[1] = *(const v8bf*)(arow + ks * 32 + 16 + 8 * half);
        }
    }

    float best[8];
    int   bidx[8];
    #pragma unroll
    for (int r = 0; r < 8; ++r) { best[r] = 3.0e38f; bidx[r] = 0; }

    // B-tile staging assignment: 16 threads per codebook row, 48 bf16 (96B).
    const int srow = tid >> 4;          // codebook row within tile (0..15)
    const int scol = (tid & 15) * 48;   // starting column (bf16 elements)
    // byte offsets of this thread's segment
    const unsigned lds_seg = (unsigned)(srow * D_PAD + scol) * 2u;
    const unsigned gv_seg  = (unsigned)(srow * D_DIM + scol) * 2u;

    // prologue: bring tile 0 into buffer 0
    async_copy96(lds_seg, gv_seg, Ebf);
    wait_async0();
    __syncthreads();

    for (int t = 0; t < NTILES; ++t) {
        // kick off tile t+1 into the other buffer; overlaps the WMMAs below
        if (t + 1 < NTILES) {
            const unsigned lds_off = lds_seg + (unsigned)((t + 1) & 1) * LDS_B_BYTES;
            const unsigned gvoff   = gv_seg + (unsigned)(t + 1) * (CTILE * D_DIM * 2u);
            async_copy96(lds_off, gvoff, Ebf);
        }

        const __bf16* Bbuf = Bs + (size_t)(t & 1) * LDS_B_ELEMS;
        v8f acc = {};
        #pragma unroll
        for (int ks = 0; ks < KSTEPS; ++ks) {
            // B fragment (32x16 bf16 = E^T tile): lane holds code row l15,
            // two 8-elem K-runs at ks*32 + 8*half and ks*32 + 16 + 8*half
            V16U bu;
            bu.h[0] = *(const v8bf*)(Bbuf + (size_t)l15 * D_PAD + ks * 32 + 8 * half);
            bu.h[1] = *(const v8bf*)(Bbuf + (size_t)l15 * D_PAD + ks * 32 + 16 + 8 * half);
            v16bf b = bu.v;
            acc = __builtin_amdgcn_wmma_f32_16x16x32_bf16(
                false, afrag[ks].v, false, b, (short)0, acc, false, false);
        }

        // Epilogue: score = e2[n] - 2*dot; running per-lane argmin.
        const int   ci  = t * CTILE + l15;
        const float e2n = E2s[ci];
        #pragma unroll
        for (int r = 0; r < 8; ++r) {
            float s = fmaf(-2.0f, acc[r], e2n);
            if (s < best[r] || (s == best[r] && ci < bidx[r])) {
                best[r] = s; bidx[r] = ci;
            }
        }

        // make tile t+1 visible to all waves; also fences buffer reuse
        if (t + 1 < NTILES) {
            wait_async0();
            __syncthreads();
        }
    }

    // Cross-lane min-reduction within each 16-lane half (rows differ by half).
    #pragma unroll
    for (int r = 0; r < 8; ++r) {
        #pragma unroll
        for (int m = 8; m >= 1; m >>= 1) {
            float os = __shfl_xor(best[r], m, 32);
            int   oi = __shfl_xor(bidx[r], m, 32);
            if (os < best[r] || (os == best[r] && oi < bidx[r])) {
                best[r] = os; bidx[r] = oi;
            }
        }
    }
    if (l15 == 0) {
        // lane 0 holds rows 0..7 of this wave's tile, lane 16 rows 8..15
        const int rowoff = rowbase + wave * 16 + half * 8;
        #pragma unroll
        for (int r = 0; r < 8; ++r) out_idx[rowoff + r] = bidx[r];
    }
}

// ---------------------------------------------------------------------------
// K4: gather quantized rows (forward STE value == codebook[idx]), per-row
// squared error partials (deterministic), bincount.  One block per token row.
// ---------------------------------------------------------------------------
__global__ void __launch_bounds__(256) vq_gather_kernel(
    const float* __restrict__ x, const float* __restrict__ cb,
    const int* __restrict__ idx, float* __restrict__ outQ,
    float* __restrict__ outIdxF, unsigned* __restrict__ counts,
    float* __restrict__ sse_partial)
{
    const int row = blockIdx.x;
    const int tid = threadIdx.x;
    const int k   = idx[row];
    const float* src = cb + (size_t)k * D_DIM;
    const float* xr  = x  + (size_t)row * D_DIM;
    float*       qr  = outQ + (size_t)row * D_DIM;

    float acc = 0.f;
    #pragma unroll
    for (int j = tid; j < D_DIM; j += 256) {
        float q = src[j];
        float d = q - xr[j];
        qr[j] = q;
        acc += d * d;
    }
    #pragma unroll
    for (int m = 16; m >= 1; m >>= 1) acc += __shfl_xor(acc, m, 32);
    __shared__ float ws[8];
    if ((tid & 31) == 0) ws[tid >> 5] = acc;
    __syncthreads();
    if (tid == 0) {
        float s = 0.f;
        #pragma unroll
        for (int w = 0; w < 8; ++w) s += ws[w];
        sse_partial[row] = s;
        outIdxF[row] = (float)k;
        atomicAdd(&counts[k], 1u);
    }
}

// ---------------------------------------------------------------------------
// K5: finalize scalars: loss, commitment, codebook, perplexity, usage.
// Single block; deterministic strided reductions.
// ---------------------------------------------------------------------------
__global__ void __launch_bounds__(256) vq_finalize_kernel(
    const unsigned* __restrict__ counts, const float* __restrict__ sse_partial,
    float* __restrict__ out5)
{
    const int tid = threadIdx.x;
    float H = 0.f, U = 0.f, S = 0.f;
    for (int k = tid; k < K_CB; k += 256) {
        float c = (float)counts[k];
        float p = c * (1.0f / (float)N_TOK);
        H += -p * logf(p + 1e-10f);
        U += (c > 0.f) ? 1.f : 0.f;
    }
    for (int i = tid; i < N_TOK; i += 256) S += sse_partial[i];

    #pragma unroll
    for (int m = 16; m >= 1; m >>= 1) {
        H += __shfl_xor(H, m, 32);
        U += __shfl_xor(U, m, 32);
        S += __shfl_xor(S, m, 32);
    }
    __shared__ float sh[8], su[8], ss[8];
    if ((tid & 31) == 0) { sh[tid >> 5] = H; su[tid >> 5] = U; ss[tid >> 5] = S; }
    __syncthreads();
    if (tid == 0) {
        float h = 0.f, u = 0.f, s = 0.f;
        #pragma unroll
        for (int w = 0; w < 8; ++w) { h += sh[w]; u += su[w]; s += ss[w]; }
        const float mse = s / ((float)N_TOK * (float)D_DIM);
        out5[0] = mse + 0.25f * mse;        // loss = codebook + 0.25*commitment
        out5[1] = mse;                      // commitment_loss
        out5[2] = mse;                      // codebook_loss
        out5[3] = expf(h);                  // perplexity
        out5[4] = u * (1.0f / (float)K_CB); // usage_rate
    }
}

// ---------------------------------------------------------------------------
// Host launcher
// ---------------------------------------------------------------------------
extern "C" void kernel_launch(void* const* d_in, const int* in_sizes, int n_in,
                              void* d_out, int out_size, void* d_ws, size_t ws_size,
                              hipStream_t stream) {
    const float* d_x  = (const float*)d_in[0];   // [N, D]
    const float* d_cb = (const float*)d_in[1];   // [K, D]

    // workspace layout (bytes)
    char* ws = (char*)d_ws;
    __bf16*   x_bf   = (__bf16*)(ws);                               // N*D*2
    __bf16*   cb_bf  = (__bf16*)(ws + (size_t)N_TOK * D_DIM * 2);   // K*D*2
    float*    e2     = (float*)(ws + (size_t)N_TOK * D_DIM * 2
                                   + (size_t)K_CB * D_DIM * 2);     // K*4
    int*      idx    = (int*)((char*)e2 + (size_t)K_CB * 4);        // N*4
    unsigned* counts = (unsigned*)((char*)idx + (size_t)N_TOK * 4); // K*4
    float*    ssep   = (float*)((char*)counts + (size_t)K_CB * 4);  // N*4

    float* outQ    = (float*)d_out;
    float* outIdxF = outQ + (size_t)N_TOK * D_DIM;
    float* out5    = outIdxF + N_TOK;

    // zero bincount (stream op, graph-capturable)
    hipMemsetAsync(counts, 0, (size_t)K_CB * 4, stream);

    // K1: codebook -> bf16 + e2
    vq_prep_codebook<<<K_CB, 128, 0, stream>>>(d_cb, cb_bf, e2);
    // K2: inputs -> bf16
    {
        int n4 = N_TOK * D_DIM / 4;
        vq_prep_inputs<<<(n4 + 255) / 256, 256, 0, stream>>>(d_x, x_bf, n4);
    }
    // K3: fused WMMA GEMM + argmin (async double-buffered B)
    vq_argmin_kernel<<<N_TOK / 128, 256, LDS_BYTES, stream>>>(x_bf, cb_bf, e2, idx);
    // K4: gather + per-row SSE + bincount
    vq_gather_kernel<<<N_TOK, 256, 0, stream>>>(d_x, d_cb, idx, outQ, outIdxF,
                                                counts, ssep);
    // K5: scalars
    vq_finalize_kernel<<<1, 256, 0, stream>>>(counts, ssep, out5);
}
